// GAMBlock_71116068487801
// MI455X (gfx1250) — matrix operand order repaired and checked by reading, not compiled, not run on hardware
//
#include <hip/hip_runtime.h>

typedef __attribute__((ext_vector_type(16))) _Float16 v16h;
typedef __attribute__((ext_vector_type(8)))  _Float16 v8h;
typedef __attribute__((ext_vector_type(8)))  float    v8f;

#define P_TOT   10368      // B*H*W = 2*72*72
#define SPATIAL 5184       // H*W
#define CCH     64
#define NNODE   1296
#define MTILES  648        // P_TOT/16

// ---------------------------------------------------------------------------
// Fragment helpers (CDNA5 wave32 WMMA layouts, CDNA5 ISA)
// A 16x32 f16: lane l -> row (l&15); l<16 holds K 0..7 & 16..23, l>=16 holds 8..15 & 24..31
__device__ inline v16h load_a16(const _Float16* __restrict__ p) {
    v16h a;
#pragma unroll
    for (int j = 0; j < 8; ++j) { a[j] = p[j]; a[j + 8] = p[j + 16]; }
    return a;
}
__device__ inline v16h zero16() {
    v16h a;
#pragma unroll
    for (int j = 0; j < 16; ++j) a[j] = (_Float16)0.f;
    return a;
}

// async copy of one 16-byte chunk: global -> LDS (ASYNCcnt-tracked, gfx1250)
__device__ inline void async_g2l_b128(const void* g, void* l) {
    unsigned loff = (unsigned)(size_t)l;                       // LDS byte offset
    unsigned long long ga = (unsigned long long)(size_t)g;     // 64-bit global addr
    asm volatile("global_load_async_to_lds_b128 %0, %1, off"
                 :: "v"(loff), "v"(ga) : "memory");
}
__device__ inline void wait_async0() {
    asm volatile("s_wait_asynccnt 0x0" ::: "memory");
}

// ---------------------------------------------------------------------------
// Layout / packing kernels
// ---------------------------------------------------------------------------
// x: [B,C,H,W] f32 -> xh: [B,H,W,C] f16
__global__ __launch_bounds__(256) void cast_x_kernel(const float* __restrict__ x,
                                                     _Float16* __restrict__ xh) {
    int idx = blockIdx.x * 256 + threadIdx.x;
    if (idx >= P_TOT * CCH) return;
    int c = idx & 63, p = idx >> 6;
    int b = p / SPATIAL, sp = p % SPATIAL;
    xh[idx] = (_Float16)x[((size_t)b * CCH + c) * SPATIAL + sp];
}

// generic B pack: B[K][N] (or [N][K] when transposed) f32 -> WMMA B tiles f16
// tile layout: [(kb*ntiles+nb)][lane][16], lane = (col&15) + 16*((k>>3)&1 in chunk)
__global__ __launch_bounds__(256) void pack_b_kernel(const float* __restrict__ B,
                                                     _Float16* __restrict__ Bp,
                                                     int K, int N, int transposed) {
    int idx = blockIdx.x * 256 + threadIdx.x;
    if (idx >= K * N) return;
    int j = idx & 15, lane = (idx >> 4) & 31, tile = idx >> 9;
    int nbc = N >> 4;
    int nb = tile % nbc, kb = tile / nbc;
    int col = (nb << 4) + (lane & 15);
    int k   = (kb << 5) + ((lane >> 4) << 3) + (j & 7) + ((j >> 3) << 4);
    float v = transposed ? B[(size_t)col * K + k] : B[(size_t)k * N + col];
    Bp[idx] = (_Float16)v;
}

// conv weight pack: w[Co=64][Ci=64][3][3] -> [tap][kb(2)][nt(4)][lane][16] f16
__global__ __launch_bounds__(256) void pack_conv_w_kernel(const float* __restrict__ w,
                                                          _Float16* __restrict__ wp) {
    int idx = blockIdx.x * 256 + threadIdx.x;
    if (idx >= 9 * 2 * 4 * 512) return;
    int j = idx & 15, lane = (idx >> 4) & 31, tile = idx >> 9;
    int nt = tile & 3, tkb = tile >> 2;
    int kb = tkb & 1, tap = tkb >> 1;
    int co = (nt << 4) + (lane & 15);
    int ci = (kb << 5) + ((lane >> 4) << 3) + (j & 7) + ((j >> 3) << 4);
    wp[idx] = (_Float16)w[((size_t)co * 64 + ci) * 9 + tap];
}

// fold conv-bias + eval-BN into scale/shift
__global__ void bn_prep_kernel(const float* __restrict__ bb, const float* __restrict__ g,
                               const float* __restrict__ be, const float* __restrict__ mn,
                               const float* __restrict__ vr, float* __restrict__ s,
                               float* __restrict__ t) {
    int c = threadIdx.x;
    float sc = g[c] * rsqrtf(vr[c] + 1e-5f);
    s[c] = sc;
    t[c] = be[c] + (bb[c] - mn[c]) * sc;
}

// ---------------------------------------------------------------------------
// Fused 3x3 conv + BN + ReLU, implicit-GEMM WMMA.
// Weights double-buffer staged into LDS per tap via async-to-LDS DMA;
// B fragments read from LDS (ds_load), A streamed from global.
// in: NHWC f16; out: f16 NHWC (optional) + f32 NHWC (optional).
// Grid must be exactly MTILES waves (81 blocks x 8 waves): barriers inside.
// ---------------------------------------------------------------------------
__global__ __launch_bounds__(256)
void conv3x3_wmma_kernel(const _Float16* __restrict__ in, const _Float16* __restrict__ wp,
                         const float* __restrict__ s, const float* __restrict__ t,
                         _Float16* __restrict__ outh, int ldo, int coff,
                         float* __restrict__ out32) {
    __shared__ _Float16 wbuf[2][4096];   // 2 x 8KB tap buffers
    int tid = threadIdx.x;
    int wave = tid >> 5, lane = tid & 31;
    int mtile = blockIdx.x * 8 + wave;
    int m0 = mtile << 4;
    int r = m0 + (lane & 15);
    int b = r / SPATIAL, sp = r % SPATIAL, y = sp / 72, x = sp % 72;
    int hi = lane >> 4, koff = hi * 8;
    v8f acc0 = {}, acc1 = {}, acc2 = {}, acc3 = {};

    // stage tap 0 (512 x 16B chunks, 2 per thread)
    for (int i = tid; i < 512; i += 256)
        async_g2l_b128((const char*)wp + (size_t)i * 16, (char*)&wbuf[0][0] + i * 16);

    for (int tap = 0; tap < 9; ++tap) {
        wait_async0();        // own async stores to LDS done
        __syncthreads();      // whole block's staging done -> buffer ready
        if (tap < 8) {        // stage next tap into the other buffer
            const char* gn = (const char*)(wp + (size_t)(tap + 1) * 4096);
            char* lb = (char*)&wbuf[(tap + 1) & 1][0];
            for (int i = tid; i < 512; i += 256)
                async_g2l_b128(gn + (size_t)i * 16, lb + i * 16);
        }
        const _Float16* wl = &wbuf[tap & 1][0];
        int dy = tap / 3 - 1, dx = tap % 3 - 1;
        int yy = y + dy, xx = x + dx;
        bool valid = (yy >= 0) & (yy < 72) & (xx >= 0) & (xx < 72);
        const _Float16* base = in + ((size_t)(b * 72 + yy) * 72 + xx) * 64;
#pragma unroll
        for (int kb = 0; kb < 2; ++kb) {
            v16h a = valid ? load_a16(base + kb * 32 + koff) : zero16();
            const _Float16* wb = wl + (size_t)kb * 4 * 512;
            v16h b0 = *(const v16h*)(wb + ((0 * 32 + lane) * 16));
            v16h b1 = *(const v16h*)(wb + ((1 * 32 + lane) * 16));
            v16h b2 = *(const v16h*)(wb + ((2 * 32 + lane) * 16));
            v16h b3 = *(const v16h*)(wb + ((3 * 32 + lane) * 16));
            acc0 = __builtin_amdgcn_wmma_f32_16x16x32_f16(false, a, false, b0, (short)0, acc0, false, false);
            acc1 = __builtin_amdgcn_wmma_f32_16x16x32_f16(false, a, false, b1, (short)0, acc1, false, false);
            acc2 = __builtin_amdgcn_wmma_f32_16x16x32_f16(false, a, false, b2, (short)0, acc2, false, false);
            acc3 = __builtin_amdgcn_wmma_f32_16x16x32_f16(false, a, false, b3, (short)0, acc3, false, false);
        }
        __syncthreads();      // all waves done reading this buffer (reused at tap+2)
    }

    int cbase = lane & 15;
    int r0 = m0 + hi * 8;
    v8f accs[4] = {acc0, acc1, acc2, acc3};
#pragma unroll
    for (int nt = 0; nt < 4; ++nt) {
        int c = nt * 16 + cbase;
        float sc = s[c], tc = t[c];
#pragma unroll
        for (int i = 0; i < 8; ++i) {
            float vv = accs[nt][i] * sc + tc;
            vv = vv > 0.f ? vv : 0.f;
            int rr = r0 + i;
            if (outh)  outh[(size_t)rr * ldo + coff + c] = (_Float16)vv;
            if (out32) out32[(size_t)rr * 64 + c] = vv;
        }
    }
}

// ---------------------------------------------------------------------------
// Spatial-attention branches (1x1 conv to 1ch, leaky, elementwise scale)
// ---------------------------------------------------------------------------
__global__ __launch_bounds__(256) void att1_kernel(const float* __restrict__ c1out,
                                                   const float* __restrict__ w11,
                                                   float* __restrict__ x1f) {
    int p = blockIdx.x * 256 + threadIdx.x;
    if (p >= P_TOT) return;
    const float* row = c1out + (size_t)p * 64;
    float a = 0.f;
#pragma unroll
    for (int c = 0; c < 64; ++c) a += row[c] * w11[c];
    a = a > 0.f ? a : 0.01f * a;
#pragma unroll
    for (int c = 0; c < 64; ++c) x1f[(size_t)p * 64 + c] = row[c] * a;
}

// x2 branch: also emits block-graph-permuted f16 features and node labels v
__global__ __launch_bounds__(256) void att2_kernel(const float* __restrict__ c1out,
                                                   const float* __restrict__ w21,
                                                   _Float16* __restrict__ feat1,
                                                   float* __restrict__ vlab) {
    int p = blockIdx.x * 256 + threadIdx.x;
    if (p >= P_TOT) return;
    const float* row = c1out + (size_t)p * 64;
    float a = 0.f;
#pragma unroll
    for (int c = 0; c < 64; ++c) a += row[c] * w21[c];
    a = a > 0.f ? a : 0.01f * a;
    int b = p / SPATIAL, sp = p % SPATIAL, y = sp / 72, x = sp % 72;
    int xi = y / 36, sh = y % 36, yi = x / 36, sw = x % 36;
    int g = b * 4 + xi * 2 + yi;
    size_t r = (size_t)g * NNODE + sh * 36 + sw;
    float sum = 0.f;
#pragma unroll
    for (int c = 0; c < 64; ++c) {
        float vv = row[c] * a;
        feat1[r * 64 + c] = (_Float16)vv;
        sum += vv;
    }
    vlab[r] = sum * (1.f / 64.f);
}

// channel attention: per-(b,c) spatial max + mean
__global__ __launch_bounds__(256) void pool_kernel(const float* __restrict__ x1f,
                                                   float* __restrict__ pooled) {
    int b = blockIdx.x >> 6, c = blockIdx.x & 63;
    float mx = -1e30f, sm = 0.f;
    for (int sp = threadIdx.x; sp < SPATIAL; sp += 256) {
        float v = x1f[((size_t)b * SPATIAL + sp) * 64 + c];
        mx = fmaxf(mx, v);
        sm += v;
    }
    __shared__ float smx[256], ssm[256];
    smx[threadIdx.x] = mx; ssm[threadIdx.x] = sm;
    __syncthreads();
    for (int o = 128; o > 0; o >>= 1) {
        if (threadIdx.x < o) {
            smx[threadIdx.x] = fmaxf(smx[threadIdx.x], smx[threadIdx.x + o]);
            ssm[threadIdx.x] += ssm[threadIdx.x + o];
        }
        __syncthreads();
    }
    if (threadIdx.x == 0) pooled[b * 64 + c] = smx[0] + ssm[0] * (1.f / SPATIAL);
}

__global__ void mlp_kernel(const float* __restrict__ pooled, const float* __restrict__ w1,
                           const float* __restrict__ b1, const float* __restrict__ w2,
                           const float* __restrict__ b2, float* __restrict__ sCA) {
    int b = blockIdx.x, tid = threadIdx.x;
    __shared__ float hid[16], pl[64];
    pl[tid] = pooled[b * 64 + tid];
    __syncthreads();
    if (tid < 16) {
        float a = b1[tid];
#pragma unroll
        for (int c = 0; c < 64; ++c) a += pl[c] * w1[tid * 64 + c];
        hid[tid] = fmaxf(a, 0.f);
    }
    __syncthreads();
    float a = b2[tid];
#pragma unroll
    for (int j = 0; j < 16; ++j) a += hid[j] * w2[tid * 16 + j];
    sCA[b * 64 + tid] = 1.f / (1.f + __expf(-a));
}

__global__ __launch_bounds__(256) void scale_kernel(const float* __restrict__ x1f,
                                                    const float* __restrict__ sCA,
                                                    _Float16* __restrict__ x1s) {
    int idx = blockIdx.x * 256 + threadIdx.x;
    if (idx >= P_TOT * 64) return;
    int c = idx & 63, p = idx >> 6;
    int b = p / SPATIAL;
    x1s[idx] = (_Float16)(x1f[idx] * sCA[b * 64 + c]);
}

// ---------------------------------------------------------------------------
// GAT projection GEMM: z = feat @ W (f16 WMMA, f32 acc).
// Epilogue: writes zT f16 [Ncol][M] and atomically accumulates el/er.
// ---------------------------------------------------------------------------
__global__ __launch_bounds__(256)
void gemm_z_kernel(const _Float16* __restrict__ A, int lda, const _Float16* __restrict__ Bp,
                   int M, int N, int K, _Float16* __restrict__ zT,
                   float* __restrict__ el, float* __restrict__ er,
                   const float* __restrict__ al, const float* __restrict__ ar,
                   int nh, int dout) {
    int wave = threadIdx.x >> 5, lane = threadIdx.x & 31;
    int ntiles = N >> 4;
    int id = blockIdx.x * 8 + wave;
    if (id >= (M >> 4) * ntiles) return;
    int mtile = id / ntiles, ntile = id % ntiles;
    int m0 = mtile << 4;
    int hi = lane >> 4, koff = hi * 8;
    const _Float16* arow = A + (size_t)(m0 + (lane & 15)) * lda;
    __builtin_prefetch(arow, 0, 3);    // global_prefetch of the A row stream
    v8f acc = {};
    int kbc = K >> 5;
    for (int kb = 0; kb < kbc; ++kb) {
        v16h a = load_a16(arow + kb * 32 + koff);
        v16h b = *(const v16h*)(Bp + ((size_t)(kb * ntiles + ntile) * 32 + lane) * 16);
        acc = __builtin_amdgcn_wmma_f32_16x16x32_f16(false, a, false, b, (short)0, acc, false, false);
    }
    int c  = (ntile << 4) + (lane & 15);
    int r0 = m0 + hi * 8;
    v8h zh;
#pragma unroll
    for (int i = 0; i < 8; ++i) zh[i] = (_Float16)acc[i];
    *(v8h*)(zT + (size_t)c * M + r0) = zh;
    int head = (ntile << 4) / dout;
    int d = c - head * dout;
    float alv = al[head * dout + d], arv = ar[head * dout + d];
    float pe[8], pr[8];
#pragma unroll
    for (int i = 0; i < 8; ++i) { pe[i] = acc[i] * alv; pr[i] = acc[i] * arv; }
#pragma unroll
    for (int m = 1; m < 16; m <<= 1) {
#pragma unroll
        for (int i = 0; i < 8; ++i) {
            pe[i] += __shfl_xor(pe[i], m, 32);
            pr[i] += __shfl_xor(pr[i], m, 32);
        }
    }
    if ((lane & 15) == 0) {
#pragma unroll
        for (int i = 0; i < 8; ++i) {
            atomicAdd(&el[(size_t)(r0 + i) * nh + head], pe[i]);
            atomicAdd(&er[(size_t)(r0 + i) * nh + head], pr[i]);
        }
    }
}

// ---------------------------------------------------------------------------
// Flash-style GAT attention: streaming softmax over j, P@Z via WMMA.
// mode 0: ELU -> featNext f16 row-major.  mode 1: (./nh) -> atomicAdd uacc f32.
// ---------------------------------------------------------------------------
template <int DT>   // dout/16
__global__ __launch_bounds__(256)
void gat_attn_kernel(const float* __restrict__ el, const float* __restrict__ er,
                     const float* __restrict__ v, const _Float16* __restrict__ zT, int M,
                     const float* __restrict__ bias, int nh, int mode,
                     _Float16* __restrict__ featNext, int ldNext, float* __restrict__ uacc) {
    const int dout = DT * 16;
    int wave = threadIdx.x >> 5, lane = threadIdx.x & 31;
    int id = blockIdx.x * 8 + wave;
    if (id >= 8 * nh * 81) return;
    int itile = id % 81, gh = id / 81;
    int h = gh % nh, g = gh / nh;
    int i0 = itile * 16;
    int lrow = lane & 15, hi = lane >> 4, koff = hi * 8;
    int irow = i0 + lrow;
    float eri = er[(size_t)(g * NNODE + irow) * nh + h];
    float vi  = v[g * NNODE + irow];
    v8f acc[DT];
#pragma unroll
    for (int t = 0; t < DT; ++t) acc[t] = (v8f){};
    float mrun = -1e30f, lrun = 0.f;

    for (int j0 = 0; j0 < NNODE; j0 += 32) {
        int jl = j0 + lane;
        float elj = 0.f, vj = 0.f;
        if (jl < NNODE) {
            elj = el[(size_t)(g * NNODE + jl) * nh + h];
            vj  = v[g * NNODE + jl];
        }
        float ev[16];
        float cmax = -1e30f;
#pragma unroll
        for (int k = 0; k < 16; ++k) {
            int jr = hi * 16 + k;
            int j = j0 + jr;
            float elv = __shfl(elj, jr, 32);
            float vv  = __shfl(vj, jr, 32);
            float e = -1e30f;
            if (j < NNODE) {
                float mult = ((vi * vv) > 0.25f ? 1.f : 0.f) + ((j == irow) ? 1.f : 0.f);
                e = eri + elv;
                e = e > 0.f ? e : 0.2f * e;
                if (mult <= 0.f) e = -1e9f;
            }
            ev[k] = e;
            cmax = fmaxf(cmax, e);
        }
        cmax = fmaxf(cmax, __shfl_xor(cmax, 16, 32));
        float newm = fmaxf(mrun, cmax);
        float resc = __expf(mrun - newm);
        // P tile (A fragment) + row sums
        v16h pa;
        float psum = 0.f;
#pragma unroll
        for (int k = 0; k < 16; ++k) {
            int jr = hi * 16 + k;
            int j = j0 + jr;
            float vv = __shfl(vj, jr, 32);
            float pk = 0.f;
            if (j < NNODE) {
                float mult = ((vi * vv) > 0.25f ? 1.f : 0.f) + ((j == irow) ? 1.f : 0.f);
                pk = mult * __expf(ev[k] - newm);
            }
            pa[k] = (_Float16)pk;
            psum += pk;
        }
        psum += __shfl_xor(psum, 16, 32);
        lrun = lrun * resc + psum;
        mrun = newm;
        // rescale accumulators (factor per row, rows live in lanes 0..15)
        int srcbase = (lane & 16) >> 1;
#pragma unroll
        for (int i = 0; i < 8; ++i) {
            float fi = __shfl(resc, srcbase + i, 32);
#pragma unroll
            for (int t = 0; t < DT; ++t) acc[t][i] *= fi;
        }
        // P @ Z
        int jstart = j0 + hi * 16;
        bool bval = jstart < NNODE;
#pragma unroll
        for (int t = 0; t < DT; ++t) {
            v16h bf = bval
                ? *(const v16h*)(zT + (size_t)(h * dout + t * 16 + lrow) * M + g * NNODE + jstart)
                : zero16();
            acc[t] = __builtin_amdgcn_wmma_f32_16x16x32_f16(false, pa, false, bf, (short)0, acc[t], false, false);
        }
    }

    float invl = 1.f / lrun;
    int srcbase = (lane & 16) >> 1;
    float invnh = 1.f / (float)nh;
#pragma unroll
    for (int i = 0; i < 8; ++i) {
        float fi = __shfl(invl, srcbase + i, 32);
        int node = i0 + i + 8 * hi;
#pragma unroll
        for (int t = 0; t < DT; ++t) {
            int c = t * 16 + lrow;
            float val = acc[t][i] * fi + bias[h * dout + c];
            if (mode == 0) {
                val = val > 0.f ? val : (__expf(val) - 1.f);   // ELU
                featNext[(size_t)(g * NNODE + node) * ldNext + h * dout + c] = (_Float16)val;
            } else {
                atomicAdd(&uacc[(size_t)(g * NNODE + node) * 64 + c], val * invnh);
            }
        }
    }
}

__global__ __launch_bounds__(256) void cast_u_kernel(const float* __restrict__ uacc,
                                                     _Float16* __restrict__ uh) {
    int idx = blockIdx.x * 256 + threadIdx.x;
    if (idx >= P_TOT * 64) return;
    uh[idx] = (_Float16)uacc[idx];
}

// ---------------------------------------------------------------------------
// amend 1x1 conv (K=128 -> 64) + leaky, writes NCHW f32 output
// ---------------------------------------------------------------------------
__global__ __launch_bounds__(256)
void gemm_amend_kernel(const _Float16* __restrict__ A, const _Float16* __restrict__ Bp,
                       float* __restrict__ out) {
    int wave = threadIdx.x >> 5, lane = threadIdx.x & 31;
    int id = blockIdx.x * 8 + wave;
    if (id >= MTILES * 4) return;
    int mtile = id >> 2, ntile = id & 3;
    int m0 = mtile << 4, hi = lane >> 4, koff = hi * 8;
    const _Float16* arow = A + (size_t)(m0 + (lane & 15)) * 128;
    v8f acc = {};
#pragma unroll
    for (int kb = 0; kb < 4; ++kb) {
        v16h a = load_a16(arow + kb * 32 + koff);
        v16h b = *(const v16h*)(Bp + ((size_t)(kb * 4 + ntile) * 32 + lane) * 16);
        acc = __builtin_amdgcn_wmma_f32_16x16x32_f16(false, a, false, b, (short)0, acc, false, false);
    }
    int c = (ntile << 4) + (lane & 15);
    int r0 = m0 + hi * 8;
    int b = r0 / SPATIAL, sp = r0 % SPATIAL;
    v8f o;
#pragma unroll
    for (int i = 0; i < 8; ++i) {
        float vv = acc[i];
        o[i] = vv > 0.f ? vv : 0.01f * vv;
    }
    *(v8f*)(out + ((size_t)b * 64 + c) * SPATIAL + sp) = o;
}

// ---------------------------------------------------------------------------
// Host launcher
// ---------------------------------------------------------------------------
// setup_inputs() dict-insertion-order indices
enum {
    IX = 0,
    C1_W1 = 1, C1_B1, C1_G1, C1_BE1, C1_M1, C1_V1, C1_W2, C1_B2, C1_G2, C1_BE2, C1_M2, C1_V2,
    AM11 = 13, MLP_W1, MLP_B1, MLP_W2, MLP_B2,
    C12_W1 = 18, C12_B1, C12_G1, C12_BE1, C12_M1, C12_V1, C12_W2, C12_B2, C12_G2, C12_BE2, C12_M2, C12_V2,
    AM21 = 30,
    G1_W = 31, G1_AL, G1_AR, G1_B,
    G2_W = 35, G2_AL, G2_AR, G2_B,
    G3_W = 39, G3_AL, G3_AR, G3_B,
    C22_W1 = 43, C22_B1, C22_G1, C22_BE1, C22_M1, C22_V1, C22_W2, C22_B2, C22_G2, C22_BE2, C22_M2, C22_V2,
    AMEND = 55
};

extern "C" void kernel_launch(void* const* d_in, const int* in_sizes, int n_in,
                              void* d_out, int out_size, void* d_ws, size_t ws_size,
                              hipStream_t stream) {
    const float* fin[64];
    for (int i = 0; i < n_in && i < 64; ++i) fin[i] = (const float*)d_in[i];
    float* out = (float*)d_out;

    // workspace carve
    char* wsb = (char*)d_ws;
    size_t off = 0;
    auto alloc = [&](size_t bytes) -> void* {
        off = (off + 255) & ~(size_t)255;
        void* p = wsb + off;
        off += bytes;
        return p;
    };
    _Float16* xh    = (_Float16*)alloc((size_t)P_TOT * 64 * 2);
    _Float16* mid   = (_Float16*)alloc((size_t)P_TOT * 64 * 2);
    float*    c1out = (float*)alloc((size_t)P_TOT * 64 * 4);
    float*    x1f   = (float*)alloc((size_t)P_TOT * 64 * 4);
    _Float16* x1s   = (_Float16*)alloc((size_t)P_TOT * 64 * 2);
    _Float16* feat1 = (_Float16*)alloc((size_t)P_TOT * 64 * 2);
    _Float16* feat2 = (_Float16*)alloc((size_t)P_TOT * 384 * 2);
    _Float16* feat3 = (_Float16*)alloc((size_t)P_TOT * 640 * 2);
    _Float16* zT    = (_Float16*)alloc((size_t)640 * P_TOT * 2 + 256);
    float*    el    = (float*)alloc((size_t)P_TOT * 5 * 4);
    float*    er    = (float*)alloc((size_t)P_TOT * 5 * 4);
    float*    vlab  = (float*)alloc((size_t)P_TOT * 4);
    float*    uacc  = (float*)alloc((size_t)P_TOT * 64 * 4);
    _Float16* uh    = (_Float16*)alloc((size_t)P_TOT * 64 * 2);
    _Float16* cat   = (_Float16*)alloc((size_t)P_TOT * 128 * 2);
    float*    pooled= (float*)alloc(128 * 4);
    float*    sCA   = (float*)alloc(128 * 4);
    _Float16* cw[6];
    for (int i = 0; i < 6; ++i) cw[i] = (_Float16*)alloc((size_t)9 * 64 * 64 * 2);
    _Float16* gw1 = (_Float16*)alloc((size_t)64 * 384 * 2);
    _Float16* gw2 = (_Float16*)alloc((size_t)384 * 640 * 2);
    _Float16* gw3 = (_Float16*)alloc((size_t)640 * 192 * 2);
    _Float16* aw  = (_Float16*)alloc((size_t)128 * 64 * 2);
    float* bns = (float*)alloc(6 * 64 * 4);
    float* bnt = (float*)alloc(6 * 64 * 4);
    if (off > ws_size) return;

    // ---- packing / prep ----
    cast_x_kernel<<<(P_TOT * 64 + 255) / 256, 256, 0, stream>>>(fin[IX], xh);
    const int convW[6]  = {C1_W1, C1_W2, C12_W1, C12_W2, C22_W1, C22_W2};
    const int convB[6]  = {C1_B1, C1_B2, C12_B1, C12_B2, C22_B1, C22_B2};
    const int convG[6]  = {C1_G1, C1_G2, C12_G1, C12_G2, C22_G1, C22_G2};
    const int convBE[6] = {C1_BE1, C1_BE2, C12_BE1, C12_BE2, C22_BE1, C22_BE2};
    const int convM[6]  = {C1_M1, C1_M2, C12_M1, C12_M2, C22_M1, C22_M2};
    const int convV[6]  = {C1_V1, C1_V2, C12_V1, C12_V2, C22_V1, C22_V2};
    for (int i = 0; i < 6; ++i) {
        pack_conv_w_kernel<<<144, 256, 0, stream>>>(fin[convW[i]], cw[i]);
        bn_prep_kernel<<<1, 64, 0, stream>>>(fin[convB[i]], fin[convG[i]], fin[convBE[i]],
                                             fin[convM[i]], fin[convV[i]],
                                             bns + i * 64, bnt + i * 64);
    }
    pack_b_kernel<<<(64 * 384 + 255) / 256, 256, 0, stream>>>(fin[G1_W], gw1, 64, 384, 0);
    pack_b_kernel<<<(384 * 640 + 255) / 256, 256, 0, stream>>>(fin[G2_W], gw2, 384, 640, 0);
    pack_b_kernel<<<(640 * 192 + 255) / 256, 256, 0, stream>>>(fin[G3_W], gw3, 640, 192, 0);
    pack_b_kernel<<<(128 * 64 + 255) / 256, 256, 0, stream>>>(fin[AMEND], aw, 128, 64, 1);

    // ---- conv1 ----
    conv3x3_wmma_kernel<<<81, 256, 0, stream>>>(xh, cw[0], bns + 0, bnt + 0, mid, 64, 0, nullptr);
    conv3x3_wmma_kernel<<<81, 256, 0, stream>>>(mid, cw[1], bns + 64, bnt + 64, nullptr, 64, 0, c1out);

    // ---- spatial + channel attention branch 1 ----
    att1_kernel<<<(P_TOT + 255) / 256, 256, 0, stream>>>(c1out, fin[AM11], x1f);
    pool_kernel<<<128, 256, 0, stream>>>(x1f, pooled);
    mlp_kernel<<<2, 64, 0, stream>>>(pooled, fin[MLP_W1], fin[MLP_B1], fin[MLP_W2], fin[MLP_B2], sCA);
    scale_kernel<<<(P_TOT * 64 + 255) / 256, 256, 0, stream>>>(x1f, sCA, x1s);
    // conv12 -> cat channels [0,64)
    conv3x3_wmma_kernel<<<81, 256, 0, stream>>>(x1s, cw[2], bns + 128, bnt + 128, mid, 64, 0, nullptr);
    conv3x3_wmma_kernel<<<81, 256, 0, stream>>>(mid, cw[3], bns + 192, bnt + 192, cat, 128, 0, nullptr);

    // ---- graph branch ----
    att2_kernel<<<(P_TOT + 255) / 256, 256, 0, stream>>>(c1out, fin[AM21], feat1, vlab);

    // GAT layer 1: Fin=64, nh=3, dout=128
    hipMemsetAsync(el, 0, (size_t)P_TOT * 3 * 4, stream);
    hipMemsetAsync(er, 0, (size_t)P_TOT * 3 * 4, stream);
    gemm_z_kernel<<<(MTILES * 24 + 7) / 8, 256, 0, stream>>>(feat1, 64, gw1, P_TOT, 384, 64,
                                                             zT, el, er, fin[G1_AL], fin[G1_AR], 3, 128);
    gat_attn_kernel<8><<<(8 * 3 * 81 + 7) / 8, 256, 0, stream>>>(el, er, vlab, zT, P_TOT,
                                                                 fin[G1_B], 3, 0, feat2, 384, nullptr);
    // GAT layer 2: Fin=384, nh=5, dout=128
    hipMemsetAsync(el, 0, (size_t)P_TOT * 5 * 4, stream);
    hipMemsetAsync(er, 0, (size_t)P_TOT * 5 * 4, stream);
    gemm_z_kernel<<<(MTILES * 40 + 7) / 8, 256, 0, stream>>>(feat2, 384, gw2, P_TOT, 640, 384,
                                                             zT, el, er, fin[G2_AL], fin[G2_AR], 5, 128);
    gat_attn_kernel<8><<<(8 * 5 * 81 + 7) / 8, 256, 0, stream>>>(el, er, vlab, zT, P_TOT,
                                                                 fin[G2_B], 5, 0, feat3, 640, nullptr);
    // GAT layer 3: Fin=640, nh=3, dout=64, mean heads
    hipMemsetAsync(el, 0, (size_t)P_TOT * 3 * 4, stream);
    hipMemsetAsync(er, 0, (size_t)P_TOT * 3 * 4, stream);
    hipMemsetAsync(uacc, 0, (size_t)P_TOT * 64 * 4, stream);
    gemm_z_kernel<<<(MTILES * 12 + 7) / 8, 256, 0, stream>>>(feat3, 640, gw3, P_TOT, 192, 640,
                                                             zT, el, er, fin[G3_AL], fin[G3_AR], 3, 64);
    gat_attn_kernel<4><<<(8 * 3 * 81 + 7) / 8, 256, 0, stream>>>(el, er, vlab, zT, P_TOT,
                                                                 fin[G3_B], 3, 1, nullptr, 0, uacc);
    cast_u_kernel<<<(P_TOT * 64 + 255) / 256, 256, 0, stream>>>(uacc, uh);

    // conv22 -> cat channels [64,128)
    conv3x3_wmma_kernel<<<81, 256, 0, stream>>>(uh, cw[4], bns + 256, bnt + 256, mid, 64, 0, nullptr);
    conv3x3_wmma_kernel<<<81, 256, 0, stream>>>(mid, cw[5], bns + 320, bnt + 320, cat, 128, 64, nullptr);

    // ---- amend 1x1 conv + leaky -> NCHW f32 output ----
    gemm_amend_kernel<<<(MTILES * 4 + 7) / 8, 256, 0, stream>>>(cat, aw, out);
}
